// ValueNetwork_26010321944671
// MI455X (gfx1250) — matrix operand order repaired and checked by reading, not compiled, not run on hardware
//
#include <hip/hip_runtime.h>

// ---------------- types ----------------
typedef __attribute__((ext_vector_type(16))) _Float16 v16h;
typedef __attribute__((ext_vector_type(8)))  _Float16 v8h;
typedef __attribute__((ext_vector_type(8)))  float    v8f;

union V16H { v16h v; v8h h[2]; };

// ---------------- problem constants ----------------
// B=16384, N=20, D=13; 16 items/block -> 320 rows -> 10 M-chunks of 32 rows (2 tiles)
#define ITEMS_PER_BLK 16
#define ROWS_PER_BLK  320
#define NCHUNKS       10
#define BLK_THREADS   128
#define GRID_BLKS     1024   // 16384/16

// LDS carve (halves for f16 regions, floats for f32 regions).
// Strides exploit zero K-padding in packed weights: A-fragment over-reads past a
// row's valid width hit zero weight rows, so narrow strides are safe (LDS is
// fully zeroed at start; no NaNs can reach a WMMA).
#define XH_STR  16
#define H1_STR  112
#define SCR_STR 160
#define XH_H   (320*XH_STR)   // x, valid K=13
#define H1_H   (320*H1_STR)   // h1, valid 100
#define G_H    (16*H1_STR)    // per-item mean g
#define M_H    (16*64)        // head input m (56 -> 64)
#define T2_H   (16*160)       // head intermediates
#define T3_H   (16*H1_STR)
#define T4_H   (16*H1_STR)
#define SCR_H  (4*32*SCR_STR) // per-wave scratch: 32 rows x 160
#define HALVES_TOTAL (XH_H+H1_H+G_H+M_H+T2_H+T3_H+T4_H+SCR_H)
#define GQ_F   (16*112)
#define SC_F   (320)
#define AT_F   (320)
#define WF_F   (16*64)
#define FLOATS_TOTAL (GQ_F+SC_F+AT_F+WF_F)
#define SMEM_BYTES (HALVES_TOTAL*2 + FLOATS_TOTAL*4)

// ---------------- packed-weight sizes (halves = KC*Np*32) ----------------
// layout: idx = ((kc*Np + n)*2 + kgroup)*16 + kk   (lane-contiguous B fragments)
#define PK_W1A  (1*160*32)
#define PK_W1B  (5*112*32)
#define PK_WA0T (4*112*32)
#define PK_WA0B (4*112*32)
#define PK_WA1  (4*112*32)
#define PK_WA2  (4*16*32)
#define PK_W2A  (4*112*32)
#define PK_W2B  (4*64*32)
#define PK_W3A  (2*160*32)
#define PK_W3B  (5*112*32)
#define PK_W3C  (4*112*32)
#define PK_W3D  (4*16*32)

// ---------------- weight pack kernel ----------------
__global__ void pack_weight(const float* __restrict__ w, _Float16* __restrict__ dst,
                            int srcK, int srcN, int srcStride, int rowOff,
                            int KC, int Np) {
    int total = KC * Np * 32;
    for (int idx = blockIdx.x * blockDim.x + threadIdx.x; idx < total;
         idx += gridDim.x * blockDim.x) {
        int kc = idx / (Np * 32);
        int r  = idx % (Np * 32);
        int n  = r / 32;
        int q  = r % 32;
        int k  = kc * 32 + (q >> 4) * 8 + ((q >> 3) & 1) * 16 + (q & 7);
        float v = (k < srcK && n < srcN) ? w[(size_t)(k + rowOff) * srcStride + n] : 0.0f;
        dst[idx] = (_Float16)v;
    }
}

// ---------------- fragment helpers ----------------
__device__ __forceinline__ v16h load_afrag(const _Float16* base, int strideH, int kc, int lane) {
    // 16-bit A 16x32 layout: lanes 0-15 hold row m, K {0..7,16..23}; lanes 16-31: K {8..15,24..31}
    int m = lane & 15, g = lane >> 4;
    const _Float16* p = base + (size_t)m * strideH + kc * 32 + g * 8;
    V16H u;
    u.h[0] = *(const v8h*)(p);
    u.h[1] = *(const v8h*)(p + 16);
    return u.v;
}

__device__ __forceinline__ v16h load_bfrag(const _Float16* wpk, int Np, int kc, int ncol0, int lane) {
    // packed so each lane reads 16 contiguous halves for (col, kgroup)
    const _Float16* p = wpk + ((size_t)((kc * Np + ncol0 + (lane & 15)) * 2 + (lane >> 4))) * 16;
    V16H u;
    u.h[0] = *(const v8h*)(p);
    u.h[1] = *(const v8h*)(p + 8);
    return u.v;
}

__device__ __forceinline__ v8f wmma_step(v16h a, v16h b, v8f c) {
    return __builtin_amdgcn_wmma_f32_16x16x32_f16(false, a, false, b, (short)0, c, false, false);
}

// MT-way M-register-blocked tile GEMM: one B-fragment load feeds MT WMMA ops.
template <int KC, int NT, int MT>
__device__ __forceinline__ void gemm_tile_m(const _Float16* A, int astride,
                                            const _Float16* W, int Np,
                                            v8f (&acc)[MT][NT], int lane) {
#pragma unroll
    for (int kc = 0; kc < KC; ++kc) {
        v16h a[MT];
#pragma unroll
        for (int mt = 0; mt < MT; ++mt)
            a[mt] = load_afrag(A + (size_t)mt * 16 * astride, astride, kc, lane);
#pragma unroll
        for (int nt = 0; nt < NT; ++nt) {
            v16h b = load_bfrag(W, Np, kc, nt * 16, lane);
#pragma unroll
            for (int mt = 0; mt < MT; ++mt)
                acc[mt][nt] = wmma_step(a[mt], b, acc[mt][nt]);
        }
    }
}

template <int NT, int MT>
__device__ __forceinline__ void store_bias_relu_m(v8f (&acc)[MT][NT], _Float16* dst, int strideH,
                                                  const float* bias, int nbias, int lane,
                                                  bool relu) {
    int col_l = lane & 15;
    int rofs  = (lane >> 4) * 8;
#pragma unroll
    for (int nt = 0; nt < NT; ++nt) {
        int col  = nt * 16 + col_l;
        float bv = (col < nbias) ? bias[col] : 0.0f;
#pragma unroll
        for (int mt = 0; mt < MT; ++mt) {
#pragma unroll
            for (int j = 0; j < 8; ++j) {
                float v = acc[mt][nt][j] + bv;
                if (relu) v = fmaxf(v, 0.0f);
                dst[(size_t)(mt * 16 + j + rofs) * strideH + col] = (_Float16)v;
            }
        }
    }
}

// ---------------- fused network kernel ----------------
__global__ __launch_bounds__(BLK_THREADS, 1)
void value_net_kernel(const float* __restrict__ x,
                      const float* __restrict__ b1a, const float* __restrict__ b1b,
                      const float* __restrict__ ba0, const float* __restrict__ ba1,
                      const float* __restrict__ ba2,
                      const float* __restrict__ b2a, const float* __restrict__ b2b,
                      const float* __restrict__ b3a, const float* __restrict__ b3b,
                      const float* __restrict__ b3c, const float* __restrict__ b3d,
                      const _Float16* __restrict__ pw1a, const _Float16* __restrict__ pw1b,
                      const _Float16* __restrict__ pwa0t, const _Float16* __restrict__ pwa0b,
                      const _Float16* __restrict__ pwa1, const _Float16* __restrict__ pwa2,
                      const _Float16* __restrict__ pw2a, const _Float16* __restrict__ pw2b,
                      const _Float16* __restrict__ pw3a, const _Float16* __restrict__ pw3b,
                      const _Float16* __restrict__ pw3c, const _Float16* __restrict__ pw3d,
                      float* __restrict__ out) {
    extern __shared__ __align__(16) char smem_raw[];
    _Float16* xh   = (_Float16*)smem_raw;      // [320][16]
    _Float16* h1   = xh   + XH_H;              // [320][112]
    _Float16* gbuf = h1   + H1_H;              // [16][112]
    _Float16* mbuf = gbuf + G_H;               // [16][64]
    _Float16* t2   = mbuf + M_H;               // [16][160]
    _Float16* t3   = t2   + T2_H;              // [16][112]
    _Float16* t4   = t3   + T3_H;              // [16][112]
    _Float16* scr  = t4   + T4_H;              // [4][32][160]
    float* gq     = (float*)(scr + SCR_H);     // [16][112]
    float* scores = gq + GQ_F;                 // [320]
    float* attn   = scores + SC_F;             // [320]
    float* wf     = attn + AT_F;               // [16][64]

    const int tid  = threadIdx.x;
    const int wave = tid >> 5;
    const int lane = tid & 31;
    const int col_l = lane & 15;
    const int rofs  = (lane >> 4) * 8;
    _Float16* myscr = scr + wave * (32 * SCR_STR);

    // zero all LDS (pads/over-reads must be finite zeros, never NaN, for WMMA)
    for (int i = tid; i < SMEM_BYTES / 4; i += BLK_THREADS)
        ((unsigned int*)smem_raw)[i] = 0u;
    __syncthreads();

    // stage x -> f16 LDS (valid K=13; K 13..31 map to zero weight rows)
    const size_t xbase = (size_t)blockIdx.x * ITEMS_PER_BLK * 20 * 13;
    for (int i = tid; i < ROWS_PER_BLK * 13; i += BLK_THREADS) {
        int r = i / 13, c = i % 13;
        xh[r * XH_STR + c] = (_Float16)x[xbase + i];
    }
    __syncthreads();

    v8f z8 = {};

    // ---- entity MLP: h1 = relu(relu(x@w1a+b1a)@w1b+b1b), 32 rows per chunk ----
    for (int c = wave; c < NCHUNKS; c += 4) {
        int rows0 = c * 32;
        v8f acc[2][10];
#pragma unroll
        for (int mt = 0; mt < 2; ++mt)
#pragma unroll
            for (int i = 0; i < 10; ++i) acc[mt][i] = z8;
        gemm_tile_m<1, 10, 2>(xh + (size_t)rows0 * XH_STR, XH_STR, pw1a, 160, acc, lane);
        store_bias_relu_m<10, 2>(acc, myscr, SCR_STR, b1a, 150, lane, true);

        v8f acc2[2][7];
#pragma unroll
        for (int mt = 0; mt < 2; ++mt)
#pragma unroll
            for (int i = 0; i < 7; ++i) acc2[mt][i] = z8;
        gemm_tile_m<5, 7, 2>(myscr, SCR_STR, pw1b, 112, acc2, lane);
        store_bias_relu_m<7, 2>(acc2, h1 + (size_t)rows0 * H1_STR, H1_STR, b1b, 100, lane, true);
    }
    __syncthreads();

    // ---- g = mean over entities ----
    for (int i = tid; i < 16 * 100; i += BLK_THREADS) {
        int it = i / 100, fc = i % 100;
        float s = 0.0f;
        for (int e = 0; e < 20; ++e) s += (float)h1[(it * 20 + e) * H1_STR + fc];
        gbuf[it * H1_STR + fc] = (_Float16)(s * (1.0f / 20.0f));
    }
    __syncthreads();

    // ---- gq = g @ wa0[100:200]  (per-item, one 16-row tile, wave 0) ----
    if (wave == 0) {
        v8f acc[1][7];
#pragma unroll
        for (int i = 0; i < 7; ++i) acc[0][i] = z8;
        gemm_tile_m<4, 7, 1>(gbuf, H1_STR, pwa0b, 112, acc, lane);
#pragma unroll
        for (int nt = 0; nt < 7; ++nt)
#pragma unroll
            for (int j = 0; j < 8; ++j)
                gq[(j + rofs) * 112 + nt * 16 + col_l] = acc[0][nt][j];
    }
    __syncthreads();

    // ---- attention scores ----
    for (int c = wave; c < NCHUNKS; c += 4) {
        int rows0 = c * 32;
        v8f acc[2][7];
#pragma unroll
        for (int mt = 0; mt < 2; ++mt)
#pragma unroll
            for (int i = 0; i < 7; ++i) acc[mt][i] = z8;
        gemm_tile_m<4, 7, 2>(h1 + (size_t)rows0 * H1_STR, H1_STR, pwa0t, 112, acc, lane);
        // s1 = relu(h1@wa0_top + gq[item] + ba0)
#pragma unroll
        for (int nt = 0; nt < 7; ++nt) {
            int col = nt * 16 + col_l;
            float bb = (col < 100) ? ba0[col] : 0.0f;
#pragma unroll
            for (int mt = 0; mt < 2; ++mt) {
#pragma unroll
                for (int j = 0; j < 8; ++j) {
                    int row = rows0 + mt * 16 + j + rofs;
                    int it  = row / 20;
                    float v = acc[mt][nt][j] + bb + gq[it * 112 + col];
                    myscr[(mt * 16 + j + rofs) * SCR_STR + col] = (_Float16)fmaxf(v, 0.0f);
                }
            }
        }
        v8f a2[2][7];
#pragma unroll
        for (int mt = 0; mt < 2; ++mt)
#pragma unroll
            for (int i = 0; i < 7; ++i) a2[mt][i] = z8;
        gemm_tile_m<4, 7, 2>(myscr, SCR_STR, pwa1, 112, a2, lane);
        store_bias_relu_m<7, 2>(a2, myscr, SCR_STR, ba1, 100, lane, true);

        v8f a3[2][1];
        a3[0][0] = z8; a3[1][0] = z8;
        gemm_tile_m<4, 1, 2>(myscr, SCR_STR, pwa2, 16, a3, lane);
        if (col_l == 0) {
#pragma unroll
            for (int mt = 0; mt < 2; ++mt)
#pragma unroll
                for (int j = 0; j < 8; ++j)
                    scores[rows0 + mt * 16 + j + rofs] = a3[mt][0][j] + ba2[0];
        }
    }
    __syncthreads();

    // ---- softmax over 20 entities per item ----
    if (tid < 16) {
        float mx = -3.0e38f;
        for (int e = 0; e < 20; ++e) mx = fmaxf(mx, scores[tid * 20 + e]);
        float s = 0.0f;
        for (int e = 0; e < 20; ++e) {
            float v = __expf(scores[tid * 20 + e] - mx);
            attn[tid * 20 + e] = v;
            s += v;
        }
        float inv = 1.0f / s;
        for (int e = 0; e < 20; ++e) attn[tid * 20 + e] *= inv;
    }
    __syncthreads();

    // ---- f = relu(h1@w2a+b2a)@w2b+b2b ; wf = sum_n attn*f ----
    for (int c = wave; c < NCHUNKS; c += 4) {
        int rows0 = c * 32;
        v8f acc[2][7];
#pragma unroll
        for (int mt = 0; mt < 2; ++mt)
#pragma unroll
            for (int i = 0; i < 7; ++i) acc[mt][i] = z8;
        gemm_tile_m<4, 7, 2>(h1 + (size_t)rows0 * H1_STR, H1_STR, pw2a, 112, acc, lane);
        store_bias_relu_m<7, 2>(acc, myscr, SCR_STR, b2a, 100, lane, true);

        v8f a2[2][4];
#pragma unroll
        for (int mt = 0; mt < 2; ++mt)
#pragma unroll
            for (int i = 0; i < 4; ++i) a2[mt][i] = z8;
        gemm_tile_m<4, 4, 2>(myscr, SCR_STR, pw2b, 64, a2, lane);
#pragma unroll
        for (int nt = 0; nt < 4; ++nt) {
            int col  = nt * 16 + col_l;
            float bb = (col < 50) ? b2b[col] : 0.0f;
#pragma unroll
            for (int mt = 0; mt < 2; ++mt) {
#pragma unroll
                for (int j = 0; j < 8; ++j) {
                    int row = rows0 + mt * 16 + j + rofs;
                    int it  = row / 20;
                    float fv = a2[mt][nt][j] + bb;
                    atomicAdd(&wf[it * 64 + col], fv * attn[row]);
                }
            }
        }
    }
    __syncthreads();

    // ---- build m = concat(self_state[6], wf[50]) -> [16][64] ----
    for (int i = tid; i < 16 * 56; i += BLK_THREADS) {
        int it = i / 56, c = i % 56;
        float v = (c < 6) ? x[xbase + (size_t)it * 260 + c] : wf[it * 64 + (c - 6)];
        mbuf[it * 64 + c] = (_Float16)v;
    }
    __syncthreads();

    // ---- head (one 16-item tile, wave 0) ----
    if (wave == 0) {
        {
            v8f acc[1][10];
#pragma unroll
            for (int i = 0; i < 10; ++i) acc[0][i] = z8;
            gemm_tile_m<2, 10, 1>(mbuf, 64, pw3a, 160, acc, lane);
            store_bias_relu_m<10, 1>(acc, t2, 160, b3a, 150, lane, true);
        }
        {
            v8f acc[1][7];
#pragma unroll
            for (int i = 0; i < 7; ++i) acc[0][i] = z8;
            gemm_tile_m<5, 7, 1>(t2, 160, pw3b, 112, acc, lane);
            store_bias_relu_m<7, 1>(acc, t3, H1_STR, b3b, 100, lane, true);
        }
        {
            v8f acc[1][7];
#pragma unroll
            for (int i = 0; i < 7; ++i) acc[0][i] = z8;
            gemm_tile_m<4, 7, 1>(t3, H1_STR, pw3c, 112, acc, lane);
            store_bias_relu_m<7, 1>(acc, t4, H1_STR, b3c, 100, lane, true);
        }
        {
            v8f acc[1][1];
            acc[0][0] = z8;
            gemm_tile_m<4, 1, 1>(t4, H1_STR, pw3d, 16, acc, lane);
            if (col_l == 0) {
#pragma unroll
                for (int j = 0; j < 8; ++j)
                    out[(size_t)blockIdx.x * 16 + j + rofs] = acc[0][0][j] + b3d[0];
            }
        }
    }
}

// ---------------- host launcher ----------------
extern "C" void kernel_launch(void* const* d_in, const int* in_sizes, int n_in,
                              void* d_out, int out_size, void* d_ws, size_t ws_size,
                              hipStream_t stream) {
    // inputs (setup_inputs order):
    // 0:x 1:w1a 2:b1a 3:w1b 4:b1b 5:wa0 6:ba0 7:wa1 8:ba1 9:wa2 10:ba2
    // 11:w2a 12:b2a 13:w2b 14:b2b 15:w3a 16:b3a 17:w3b 18:b3b 19:w3c 20:b3c 21:w3d 22:b3d
    const float* x   = (const float*)d_in[0];
    const float* w1a = (const float*)d_in[1];  const float* b1a = (const float*)d_in[2];
    const float* w1b = (const float*)d_in[3];  const float* b1b = (const float*)d_in[4];
    const float* wa0 = (const float*)d_in[5];  const float* ba0 = (const float*)d_in[6];
    const float* wa1 = (const float*)d_in[7];  const float* ba1 = (const float*)d_in[8];
    const float* wa2 = (const float*)d_in[9];  const float* ba2 = (const float*)d_in[10];
    const float* w2a = (const float*)d_in[11]; const float* b2a = (const float*)d_in[12];
    const float* w2b = (const float*)d_in[13]; const float* b2b = (const float*)d_in[14];
    const float* w3a = (const float*)d_in[15]; const float* b3a = (const float*)d_in[16];
    const float* w3b = (const float*)d_in[17]; const float* b3b = (const float*)d_in[18];
    const float* w3c = (const float*)d_in[19]; const float* b3c = (const float*)d_in[20];
    const float* w3d = (const float*)d_in[21]; const float* b3d = (const float*)d_in[22];
    float* out = (float*)d_out;

    // carve packed-weight buffers out of d_ws (all sizes are multiples of 32B)
    _Float16* ws = (_Float16*)d_ws;
    size_t off = 0;
    _Float16* pw1a  = ws + off; off += PK_W1A;
    _Float16* pw1b  = ws + off; off += PK_W1B;
    _Float16* pwa0t = ws + off; off += PK_WA0T;
    _Float16* pwa0b = ws + off; off += PK_WA0B;
    _Float16* pwa1  = ws + off; off += PK_WA1;
    _Float16* pwa2  = ws + off; off += PK_WA2;
    _Float16* pw2a  = ws + off; off += PK_W2A;
    _Float16* pw2b  = ws + off; off += PK_W2B;
    _Float16* pw3a  = ws + off; off += PK_W3A;
    _Float16* pw3b  = ws + off; off += PK_W3B;
    _Float16* pw3c  = ws + off; off += PK_W3C;
    _Float16* pw3d  = ws + off; off += PK_W3D;

    auto pack = [&](const float* w, _Float16* dst, int K, int Nn, int stride, int rowOff,
                    int KC, int Np) {
        int total  = KC * Np * 32;
        int blocks = (total + 255) / 256;
        pack_weight<<<blocks, 256, 0, stream>>>(w, dst, K, Nn, stride, rowOff, KC, Np);
    };
    pack(w1a, pw1a, 13, 150, 150, 0, 1, 160);
    pack(w1b, pw1b, 150, 100, 100, 0, 5, 112);
    pack(wa0, pwa0t, 100, 100, 100, 0,   4, 112);  // rows 0..99   (h1 part)
    pack(wa0, pwa0b, 100, 100, 100, 100, 4, 112);  // rows 100..199 (g part)
    pack(wa1, pwa1, 100, 100, 100, 0, 4, 112);
    pack(wa2, pwa2, 100, 1, 1, 0, 4, 16);
    pack(w2a, pw2a, 100, 100, 100, 0, 4, 112);
    pack(w2b, pw2b, 100, 50, 50, 0, 4, 64);
    pack(w3a, pw3a, 56, 150, 150, 0, 2, 160);
    pack(w3b, pw3b, 150, 100, 100, 0, 5, 112);
    pack(w3c, pw3c, 100, 100, 100, 0, 4, 112);
    pack(w3d, pw3d, 100, 1, 1, 0, 4, 16);

    // allow >64KB dynamic LDS (idempotent; safe to call every launch)
    (void)hipFuncSetAttribute((const void*)value_net_kernel,
                              hipFuncAttributeMaxDynamicSharedMemorySize, SMEM_BYTES);

    value_net_kernel<<<GRID_BLKS, BLK_THREADS, SMEM_BYTES, stream>>>(
        x, b1a, b1b, ba0, ba1, ba2, b2a, b2b, b3a, b3b, b3c, b3d,
        pw1a, pw1b, pwa0t, pwa0b, pwa1, pwa2, pw2a, pw2b, pw3a, pw3b, pw3c, pw3d,
        out);
}
